// MultiHeadAttention_43525198578201
// MI455X (gfx1250) — compile-verified, matrix operand order checked by reading, not run on hardware
//
#include <hip/hip_runtime.h>

#define BB 2
#define SS 1024
#define DD 1024
#define HH 16
#define DEPTH 64

typedef __attribute__((ext_vector_type(16))) __bf16 v16bf;
typedef __attribute__((ext_vector_type(8)))  float  v8f;
typedef int v4i __attribute__((vector_size(16)));
typedef unsigned short bf_t;   // raw bf16 storage

#define AS1 __attribute__((address_space(1)))
#define AS3 __attribute__((address_space(3)))

// ---- optional gfx1250 async global->LDS path (guarded; sync fallback) ------
#if defined(__has_builtin)
#  if __has_builtin(__builtin_amdgcn_global_load_async_to_lds_b128)
#    define HAVE_ASYNC_G2L 1
#  endif
#endif
#ifndef HAVE_ASYNC_G2L
#  define HAVE_ASYNC_G2L 0
#endif

union FragU {
  uint4 u[2];
  unsigned short s[16];
  v16bf v;
};

// fp32 -> bf16 bits, round-to-nearest-even
__device__ __forceinline__ unsigned short f2b_bits(float x) {
  unsigned int u = __float_as_uint(x);
  unsigned int r = u + 0x7FFFu + ((u >> 16) & 1u);
  return (unsigned short)(r >> 16);
}

// A/B fragment for v_wmma_f32_16x16x32_bf16 from row-major bf16 memory.
// lane L holds row (L&15); lanes 0-15 get K = k0+[0..7] and k0+[16..23],
// lanes 16-31 get K = k0+[8..15] and k0+[24..31]  (ISA 7.12.2).
__device__ __forceinline__ v16bf load_frag_bf(const bf_t* row, int k0, int lane) {
  const bf_t* p = row + k0 + ((lane >> 4) & 1) * 8;
  FragU f;
  f.u[0] = *(const uint4*)(p);
  f.u[1] = *(const uint4*)(p + 16);
  return f.v;
}

// Same fragment, sourced from fp32 memory with in-register bf16 conversion.
__device__ __forceinline__ v16bf load_frag_f32(const float* row, int k0, int lane) {
  const float* p = row + k0 + ((lane >> 4) & 1) * 8;
  FragU f;
#pragma unroll
  for (int i = 0; i < 8; ++i) f.s[i] = f2b_bits(p[i]);
#pragma unroll
  for (int i = 0; i < 8; ++i) f.s[8 + i] = f2b_bits(p[16 + i]);
  return f.v;
}

__device__ __forceinline__ void wmma_step(v8f& acc, v16bf a, v16bf b) {
  acc = __builtin_amdgcn_wmma_f32_16x16x32_bf16(false, a, false, b,
                                                (short)0, acc, false, false);
}

// 16B global -> LDS copy (async on CDNA5 toolchains that expose the builtin)
__device__ __forceinline__ void g2l_b128(bf_t* dst_lds, const bf_t* src) {
#if HAVE_ASYNC_G2L
  __builtin_amdgcn_global_load_async_to_lds_b128(
      (AS1 v4i*)(unsigned long long)(const void*)src,
      (AS3 v4i*)(unsigned int)(unsigned long long)(void*)dst_lds,
      0, 0);
#else
  *(uint4*)dst_lds = *(const uint4*)src;
#endif
}

__device__ __forceinline__ void wait_async0() {
#if HAVE_ASYNC_G2L
#  if __has_builtin(__builtin_amdgcn_s_wait_asynccnt)
  __builtin_amdgcn_s_wait_asynccnt(0);
#  else
  asm volatile("s_wait_asynccnt 0x0" ::: "memory");
#  endif
#endif
}

// ---- kernels ---------------------------------------------------------------

// W (k-major, fp32) -> Wt (n-major, bf16)
__global__ void k_cvt_wt(const float* __restrict__ W, bf_t* __restrict__ Wt) {
  int idx = blockIdx.x * blockDim.x + threadIdx.x;
  if (idx >= DD * DD) return;
  int n = idx / DD, k = idx % DD;
  Wt[idx] = f2b_bits(W[(size_t)k * DD + n]);
}

// X(BB*SS, DD) fp32 @ Wt + bias -> head-split bf16. One 16x64 strip per wave.
// vmode==0: out[b,h,s,d]; vmode==1 (for V): out[b,h,d,s]
__global__ void k_proj(const float* __restrict__ X, const bf_t* __restrict__ Wt,
                       const float* __restrict__ bias, bf_t* __restrict__ out,
                       int vmode) {
  int wave = (blockIdx.x * blockDim.x + threadIdx.x) >> 5;
  int lane = threadIdx.x & 31;
  const int NTS = DD / 64;                 // 16 strips
  int nts = wave % NTS;
  int mt  = wave / NTS;
  if (mt >= (BB * SS) / 16) return;
  const float* arow = X + ((size_t)mt * 16 + (lane & 15)) * DD;
  const bf_t* Bt = Wt + (size_t)nts * 64 * DD;
  const bf_t* brow0 = Bt + (size_t)(lane & 15) * DD;
  v8f acc[4] = {};
  for (int k0 = 0; k0 < DD; k0 += 32) {
    __builtin_prefetch(arow + k0 + 128, 0, 1);
    __builtin_prefetch(brow0 + k0 + 256, 0, 1);
    v16bf a = load_frag_f32(arow, k0, lane);
#pragma unroll
    for (int j = 0; j < 4; ++j)
      wmma_step(acc[j], a, load_frag_bf(brow0 + (size_t)j * 16 * DD, k0, lane));
  }
  int n = lane & 15, mh = (lane >> 4) * 8;
#pragma unroll
  for (int j = 0; j < 4; ++j) {
    int col = nts * 64 + j * 16 + n;
    int h = col / DEPTH, d = col % DEPTH;
    float bias_n = bias[col];
#pragma unroll
    for (int r = 0; r < 8; ++r) {
      int row = mt * 16 + mh + r;          // = b*SS + s
      int b = row / SS, s = row % SS;
      float val = acc[j][r] + bias_n;
      size_t o = (vmode == 0)
                     ? (((size_t)(b * HH + h) * SS + s) * DEPTH + d)
                     : (((size_t)(b * HH + h) * DEPTH + d) * SS + s);
      out[o] = f2b_bits(val);
    }
  }
}

// logits[b,h,q,k] = qh . kh  (K=64); one 16x64 strip per wave
__global__ void k_qk(const bf_t* __restrict__ qh, const bf_t* __restrict__ kh,
                     float* __restrict__ logits) {
  int wave = (blockIdx.x * blockDim.x + threadIdx.x) >> 5;
  int lane = threadIdx.x & 31;
  const int NS = SS / 64;                  // 16 strips
  const int QT = SS / 16;
  int kts = wave % NS;
  int qt  = (wave / NS) % QT;
  int h   = (wave / (NS * QT)) % HH;
  int b   =  wave / (NS * QT * HH);
  if (b >= BB) return;
  const bf_t* arow = qh + ((((size_t)(b * HH + h) * SS) + qt * 16) + (lane & 15)) * DEPTH;
  const bf_t* brow0 = kh + ((((size_t)(b * HH + h) * SS) + kts * 64) + (lane & 15)) * DEPTH;
  v8f acc[4] = {};
#pragma unroll
  for (int k0 = 0; k0 < DEPTH; k0 += 32) {
    v16bf a = load_frag_bf(arow, k0, lane);
#pragma unroll
    for (int j = 0; j < 4; ++j)
      wmma_step(acc[j], a, load_frag_bf(brow0 + (size_t)j * 16 * DEPTH, k0, lane));
  }
  int n = lane & 15, mh = (lane >> 4) * 8;
  float* C = logits + (((size_t)(b * HH + h) * SS + qt * 16) * SS) + kts * 64;
#pragma unroll
  for (int j = 0; j < 4; ++j)
#pragma unroll
    for (int r = 0; r < 8; ++r)
      C[(size_t)(mh + r) * SS + j * 16 + n] = acc[j][r];
}

// logits[b,h,q,k] += qh[b,h,q,:] . rel[b,q,k,:]
// per (b,q): M=16 heads (A strided by head), one 16x64 strip of k per wave
__global__ void k_rel(const bf_t* __restrict__ qh, const float* __restrict__ rel,
                      float* __restrict__ logits) {
  int wave = (blockIdx.x * blockDim.x + threadIdx.x) >> 5;
  int lane = threadIdx.x & 31;
  const int NS = SS / 64;                  // 16 strips
  int nts = wave % NS;
  int q   = (wave / NS) % SS;
  int b   =  wave / (NS * SS);
  if (b >= BB) return;
  const bf_t* arow = qh + ((size_t)b * HH * SS + q) * DEPTH +
                     (size_t)(lane & 15) * SS * DEPTH;     // row = head
  const float* brow0 = rel + (((size_t)(b * SS + q) * SS) + nts * 64 + (lane & 15)) * DEPTH;
  v8f acc[4] = {};
#pragma unroll
  for (int k0 = 0; k0 < DEPTH; k0 += 32) {
    v16bf a = load_frag_bf(arow, k0, lane);
#pragma unroll
    for (int j = 0; j < 4; ++j)
      wmma_step(acc[j], a, load_frag_f32(brow0 + (size_t)j * 16 * DEPTH, k0, lane));
  }
  int n = lane & 15, mh = (lane >> 4) * 8;
#pragma unroll
  for (int j = 0; j < 4; ++j)
#pragma unroll
    for (int r = 0; r < 8; ++r) {
      int h = mh + r;
      float* p = logits + (((size_t)(b * HH + h) * SS + q) * SS) + nts * 64 + j * 16 + n;
      *p += acc[j][r];
    }
}

// one wave per (b,h,q) row: scale + mask + softmax; fp32 attn to d_out, bf16 to ws
__global__ void k_softmax(const float* __restrict__ logits,
                          const float* __restrict__ mask,
                          float* __restrict__ attn_f32,
                          bf_t* __restrict__ attn_bf) {
  int wave = (blockIdx.x * blockDim.x + threadIdx.x) >> 5;
  int lane = threadIdx.x & 31;
  if (wave >= BB * HH * SS) return;
  int q = wave % SS;
  int b = wave / (SS * HH);
  const float* row  = logits + (size_t)wave * SS;
  const float* mrow = mask + ((size_t)b * SS + q) * SS;
  const float scale = 0.125f;  // 1/sqrt(64)
  float vals[SS / 32];
  float mx = -3.0e38f;
#pragma unroll
  for (int i = 0; i < SS / 32; ++i) {
    int j = lane + i * 32;
    float x = row[j] * scale + mrow[j] * (-1e9f);
    vals[i] = x;
    mx = fmaxf(mx, x);
  }
#pragma unroll
  for (int off = 16; off >= 1; off >>= 1)
    mx = fmaxf(mx, __shfl_xor(mx, off, 32));
  float sum = 0.f;
#pragma unroll
  for (int i = 0; i < SS / 32; ++i) {
    float e = __expf(vals[i] - mx);
    vals[i] = e;
    sum += e;
  }
#pragma unroll
  for (int off = 16; off >= 1; off >>= 1)
    sum += __shfl_xor(sum, off, 32);
  float inv = 1.0f / sum;
  float* orow = attn_f32 + (size_t)wave * SS;
  bf_t*  brow = attn_bf + (size_t)wave * SS;
#pragma unroll
  for (int i = 0; i < SS / 32; ++i) {
    int j = lane + i * 32;
    float a = vals[i] * inv;
    orow[j] = a;
    brow[j] = f2b_bits(a);
  }
}

// ctx[b,s,h*64+d] = attn[b,h,s,:] @ vht[b,h,d,:]   (K = S = 1024)
// Workgroup = 8 waves sharing one (b,h); V panel streamed through LDS in
// 128-column chunks via async global->LDS; each wave: 16 q-rows x all 64 d.
#define KC 128
#define LVS (KC + 8)   // padded LDS row stride (breaks 256B bank period)

__global__ void k_av(const bf_t* __restrict__ attn, const bf_t* __restrict__ vht,
                     bf_t* __restrict__ ctx) {
  __shared__ __align__(16) bf_t lv[DEPTH * LVS];
  int tid = threadIdx.x;
  int lane = tid & 31;
  int w = tid >> 5;                        // wave in block: 0..7
  int bh = blockIdx.x >> 3;                // b*HH + h
  int qg = blockIdx.x & 7;
  int qt = qg * 8 + w;                     // q-tile of this wave
  int b = bh / HH, h = bh % HH;
  const bf_t* arow = attn + ((((size_t)bh * SS) + qt * 16) + (lane & 15)) * SS;
  const bf_t* Vbase = vht + (size_t)bh * DEPTH * SS;
  // per-thread stage slice: 32 elements (64B) of one d-row
  int sd = tid >> 2;                       // 0..63
  int sk = (tid & 3) * 32;                 // 0,32,64,96
  v8f acc[4] = {};
  for (int k0 = 0; k0 < SS; k0 += KC) {
    const bf_t* src = Vbase + (size_t)sd * SS + k0 + sk;
    bf_t* dst = &lv[sd * LVS + sk];
    g2l_b128(dst, src);
    g2l_b128(dst + 8, src + 8);
    g2l_b128(dst + 16, src + 16);
    g2l_b128(dst + 24, src + 24);
    wait_async0();
    __syncthreads();
#pragma unroll
    for (int ks = 0; ks < KC; ks += 32) {
      v16bf a = load_frag_bf(arow, k0 + ks, lane);
#pragma unroll
      for (int j = 0; j < 4; ++j) {
        const bf_t* brow = &lv[(j * 16 + (lane & 15)) * LVS];
        wmma_step(acc[j], a, load_frag_bf(brow, ks, lane));
      }
    }
    __syncthreads();
  }
  int n = lane & 15, mh = (lane >> 4) * 8;
#pragma unroll
  for (int j = 0; j < 4; ++j)
#pragma unroll
    for (int r = 0; r < 8; ++r) {
      int s = qt * 16 + mh + r;
      ctx[((size_t)(b * SS + s) * DD) + h * DEPTH + j * 16 + n] = f2b_bits(acc[j][r]);
    }
}

// out = ctx @ Wo + bo  (fp32 result); one 16x64 strip per wave
__global__ void k_out(const bf_t* __restrict__ ctx, const bf_t* __restrict__ Wto,
                      const float* __restrict__ bo, float* __restrict__ out) {
  int wave = (blockIdx.x * blockDim.x + threadIdx.x) >> 5;
  int lane = threadIdx.x & 31;
  const int NTS = DD / 64;
  int nts = wave % NTS;
  int mt  = wave / NTS;
  if (mt >= BB * SS / 16) return;
  const bf_t* arow = ctx + ((size_t)mt * 16 + (lane & 15)) * DD;
  const bf_t* brow0 = Wto + ((size_t)nts * 64 + (lane & 15)) * DD;
  v8f acc[4] = {};
  for (int k0 = 0; k0 < DD; k0 += 32) {
    __builtin_prefetch(arow + k0 + 256, 0, 1);
    __builtin_prefetch(brow0 + k0 + 256, 0, 1);
    v16bf a = load_frag_bf(arow, k0, lane);
#pragma unroll
    for (int j = 0; j < 4; ++j)
      wmma_step(acc[j], a, load_frag_bf(brow0 + (size_t)j * 16 * DD, k0, lane));
  }
  int n = lane & 15, mh = (lane >> 4) * 8;
#pragma unroll
  for (int j = 0; j < 4; ++j) {
    float bias_n = bo[nts * 64 + j * 16 + n];
#pragma unroll
    for (int r = 0; r < 8; ++r)
      out[(size_t)(mt * 16 + mh + r) * DD + nts * 64 + j * 16 + n] = acc[j][r] + bias_n;
  }
}

// ---- launch ---------------------------------------------------------------

extern "C" void kernel_launch(void* const* d_in, const int* in_sizes, int n_in,
                              void* d_out, int out_size, void* d_ws, size_t ws_size,
                              hipStream_t stream) {
  const float* v    = (const float*)d_in[0];
  const float* k    = (const float*)d_in[1];
  const float* q    = (const float*)d_in[2];
  const float* mask = (const float*)d_in[3];
  const float* rel  = (const float*)d_in[4];
  const float* Wq   = (const float*)d_in[5];
  const float* bq   = (const float*)d_in[6];
  const float* Wk   = (const float*)d_in[7];
  const float* bk   = (const float*)d_in[8];
  const float* Wv   = (const float*)d_in[9];
  const float* bv   = (const float*)d_in[10];
  const float* Wo   = (const float*)d_in[11];
  const float* bo   = (const float*)d_in[12];

  float* out      = (float*)d_out;
  float* attn_out = out + (size_t)BB * SS * DD;

  char* w = (char*)d_ws;
  size_t off = 0;
  auto carve = [&](size_t bytes) -> char* {
    char* p = w + off;
    off += (bytes + 255) & ~(size_t)255;
    return p;
  };
  bf_t* qh    = (bf_t*)carve((size_t)BB * HH * SS * DEPTH * 2);
  bf_t* kh    = (bf_t*)carve((size_t)BB * HH * SS * DEPTH * 2);
  bf_t* vht   = (bf_t*)carve((size_t)BB * HH * SS * DEPTH * 2);
  bf_t* Wtq   = (bf_t*)carve((size_t)DD * DD * 2);
  bf_t* Wtk   = (bf_t*)carve((size_t)DD * DD * 2);
  bf_t* Wtv   = (bf_t*)carve((size_t)DD * DD * 2);
  bf_t* Wto   = (bf_t*)carve((size_t)DD * DD * 2);
  float* logits = (float*)carve((size_t)BB * HH * SS * SS * 4);
  bf_t* attnb = (bf_t*)carve((size_t)BB * HH * SS * SS * 2);
  bf_t* ctx   = (bf_t*)carve((size_t)BB * SS * DD * 2);

  dim3 blk(256);
  int cvt_blocks = (DD * DD + 255) / 256;
  k_cvt_wt<<<cvt_blocks, blk, 0, stream>>>(Wq, Wtq);
  k_cvt_wt<<<cvt_blocks, blk, 0, stream>>>(Wk, Wtk);
  k_cvt_wt<<<cvt_blocks, blk, 0, stream>>>(Wv, Wtv);
  k_cvt_wt<<<cvt_blocks, blk, 0, stream>>>(Wo, Wto);

  int proj_blocks = ((BB * SS / 16) * (DD / 64)) / 8;     // 256
  k_proj<<<proj_blocks, blk, 0, stream>>>(q, Wtq, bq, qh, 0);
  k_proj<<<proj_blocks, blk, 0, stream>>>(k, Wtk, bk, kh, 0);
  k_proj<<<proj_blocks, blk, 0, stream>>>(v, Wtv, bv, vht, 1);

  int qk_blocks = (BB * HH * (SS / 16) * (SS / 64)) / 8;  // 4096
  k_qk<<<qk_blocks, blk, 0, stream>>>(qh, kh, logits);

  int rel_blocks = (BB * SS * (SS / 64)) / 8;             // 4096
  k_rel<<<rel_blocks, blk, 0, stream>>>(qh, rel, logits);

  int sm_blocks = (BB * HH * SS) / 8;                     // 4096
  k_softmax<<<sm_blocks, blk, 0, stream>>>(logits, mask, attn_out, attnb);

  int av_blocks = BB * HH * 8;                            // 256 (8 waves each)
  k_av<<<av_blocks, blk, 0, stream>>>(attnb, vht, ctx);

  int out_blocks = ((BB * SS / 16) * (DD / 64)) / 8;      // 256
  k_out<<<out_blocks, blk, 0, stream>>>(ctx, Wto, bo, out);
}
